// MultiHeadNeighborhoodAttentionModule_6923487282294
// MI455X (gfx1250) — compile-verified, matrix operand order checked by reading, not run on hardware
//
#include <hip/hip_runtime.h>
#include <hip/hip_bf16.h>
#include <math.h>

// ---- problem dims (fixed by reference) ----
#define EMB 256
#define NH  2
#define AD  64
#define KNB 16
#define HA  128   // NH*AD
#define GIN 512   // 2*EMB

typedef __attribute__((ext_vector_type(16))) __bf16 v16bf;
typedef __attribute__((ext_vector_type(8)))  float  v8f;

union ABu { v16bf v; uint4 q[2]; };
union ABe { v16bf v; __bf16 e[16]; };

// CDNA5 async global->LDS copy path (ASYNCcnt), with safe fallback.
#if __has_builtin(__builtin_amdgcn_global_load_async_to_lds_b128) && \
    __has_builtin(__builtin_amdgcn_s_wait_asynccnt)
#define HAVE_ASYNC_LDS 1
typedef __attribute__((ext_vector_type(4))) int v4i;
typedef __attribute__((address_space(1))) v4i gv4i;   // global int4 vector
typedef __attribute__((address_space(3))) v4i lv4i;   // LDS int4 vector
#else
#define HAVE_ASYNC_LDS 0
#endif

// Convert f32 weights to bf16 in d_ws with WMMA-B-friendly [N][K] layouts.
//   wq_b : [128 (h*64+a)][256 d]        (natural layout of W_q)
//   wk_t : [2][256 d][64 a]             (transposed W_k; B for r-GEMM)
//   wg_b : [256 out][512 in]            (natural layout of W_gate)
__global__ void prep_weights(const float* __restrict__ Wq,
                             const float* __restrict__ Wk,
                             const float* __restrict__ Wg,
                             __bf16* __restrict__ wq_b,
                             __bf16* __restrict__ wk_t,
                             __bf16* __restrict__ wg_b) {
  const int idx = blockIdx.x * blockDim.x + threadIdx.x;
  if (idx < NH * AD * EMB) {
    wq_b[idx] = (__bf16)Wq[idx];
    const int h   = idx / (EMB * AD);
    const int rem = idx - h * EMB * AD;
    const int d   = rem / AD;
    const int a   = rem - d * AD;
    wk_t[idx] = (__bf16)Wk[(h * AD + a) * EMB + d];
  }
  if (idx < EMB * GIN) wg_b[idx] = (__bf16)Wg[idx];
}

// One wave (32 lanes) per block; each block owns 16 rows of the batch.
__global__ __launch_bounds__(32) void fused_attn(
    const float* __restrict__ center, const float* __restrict__ neigh,
    const float* __restrict__ conf,   const unsigned char* __restrict__ valid,
    const float* __restrict__ b_gate,
    const __bf16* __restrict__ wq_b,  const __bf16* __restrict__ wk_t,
    const __bf16* __restrict__ wg_b,  float* __restrict__ out)
{
  __shared__ __bf16 s_center[16 * EMB];   // 8 KB  center tile, bf16
  __shared__ __bf16 s_q[16 * HA];         // 4 KB  Q = center x Wq^T
  __shared__ __bf16 s_r[16 * GIN];        // 16 KB r[b][h*256+d] = Wk^T q
  __shared__ float  s_nb[KNB * EMB];      // 16 KB one row's neigh block (f32)
  __shared__ float  s_ctx[16 * EMB];      // 16 KB context (f32)
  __shared__ float  s_attn[NH * KNB];     // softmax weights for current row

  const int  lane = threadIdx.x;
  const int  half = lane >> 4;            // which 16-lane half
  const int  lm   = lane & 15;
  const long b0   = (long)blockIdx.x * 16;

  // ---- stage center tile [16,256] f32 -> bf16 LDS (coalesced b128 loads) ----
  for (int i = 0; i < 32; ++i) {
    const int    idx = i * 32 + lane;                 // float4 index, 1024 total
    const float4 cv  = ((const float4*)(center + b0 * EMB))[idx];
    __bf16* dst = s_center + idx * 4;
    dst[0] = (__bf16)cv.x; dst[1] = (__bf16)cv.y;
    dst[2] = (__bf16)cv.z; dst[3] = (__bf16)cv.w;
  }
  __syncthreads();

  // ---- Q = center x Wq^T : [16,256]x[256,128] via v_wmma_f32_16x16x32_bf16 --
  // 16-bit A layout: lane row = lane%16, k runs {k0+half*8..+7} and
  // {k0+16+half*8..+7} -> two contiguous 16B loads per operand.
  for (int nt = 0; nt < 8; ++nt) {
    v8f acc = {};
    const int n = nt * 16 + lm;
    #pragma unroll
    for (int ks = 0; ks < 8; ++ks) {
      const int k0 = ks * 32;
      ABu a, bb;
      a.q[0]  = *(const uint4*)(s_center + lm * EMB + k0 + half * 8);
      a.q[1]  = *(const uint4*)(s_center + lm * EMB + k0 + 16 + half * 8);
      bb.q[0] = *(const uint4*)(wq_b + n * EMB + k0 + half * 8);
      bb.q[1] = *(const uint4*)(wq_b + n * EMB + k0 + 16 + half * 8);
      acc = __builtin_amdgcn_wmma_f32_16x16x32_bf16(false, a.v, false, bb.v,
                                                    (short)0, acc, false, false);
    }
    #pragma unroll
    for (int i = 0; i < 8; ++i)          // C layout: M = i+8*half, N = n
      s_q[(i + 8 * half) * HA + n] = (__bf16)acc[i];
  }
  __syncthreads();

  // ---- r[h] = Q[:,h*64:+64] x Wk_t[h] : [16,64]x[64,256] per head ----------
  for (int h = 0; h < NH; ++h) {
    for (int nt = 0; nt < 16; ++nt) {
      v8f acc = {};
      const int n = nt * 16 + lm;        // output d index
      #pragma unroll
      for (int ks = 0; ks < 2; ++ks) {
        const int k0 = h * 64 + ks * 32; // Q column base (head h slice)
        ABu a, bb;
        a.q[0] = *(const uint4*)(s_q + lm * HA + k0 + half * 8);
        a.q[1] = *(const uint4*)(s_q + lm * HA + k0 + 16 + half * 8);
        const __bf16* wr = wk_t + (h * EMB + n) * AD;
        const int kk0 = ks * 32;
        bb.q[0] = *(const uint4*)(wr + kk0 + half * 8);
        bb.q[1] = *(const uint4*)(wr + kk0 + 16 + half * 8);
        acc = __builtin_amdgcn_wmma_f32_16x16x32_bf16(false, a.v, false, bb.v,
                                                      (short)0, acc, false, false);
      }
      #pragma unroll
      for (int i = 0; i < 8; ++i)
        s_r[(i + 8 * half) * GIN + h * EMB + n] = (__bf16)acc[i];
    }
  }
  __syncthreads();

  // ---- per-row: stream neigh once (async->LDS); scores; softmax; ctx -------
  for (int bs = 0; bs < 16; ++bs) {
    const long b = b0 + bs;
    if (bs < 15)  // global_prefetch_b8 warms L2 for next row's neigh block
      __builtin_prefetch(neigh + (b + 1) * KNB * EMB + (long)lane * 128, 0, 0);

#if HAVE_ASYNC_LDS
    {   // GLOBAL_LOAD_ASYNC_TO_LDS_B128: 16 KB copy bypassing VGPRs
      const float* gsrc = neigh + b * KNB * EMB;
      #pragma unroll
      for (int i = 0; i < 32; ++i) {
        const int idx = i * 32 + lane;   // int4 index, 1024 total
        __builtin_amdgcn_global_load_async_to_lds_b128(
            (gv4i*)(gsrc + idx * 4), (lv4i*)(s_nb + idx * 4), 0, 0);
      }
      __builtin_amdgcn_s_wait_asynccnt(0);
    }
#else
    for (int i = 0; i < 32; ++i) {
      const int idx = i * 32 + lane;
      ((float4*)s_nb)[idx] = ((const float4*)(neigh + b * KNB * EMB))[idx];
    }
#endif
    __syncthreads();

    {   // lane = (h = half, kk = lane%16): one score dot per lane
      const int h = half, kk = lm;
      float s = 0.f;
      const __bf16* rrow = s_r  + bs * GIN + h * EMB;
      const float*  nrow = s_nb + kk * EMB;
      for (int d = 0; d < EMB; d += 8) {
        const float4 n0 = *(const float4*)(nrow + d);
        const float4 n1 = *(const float4*)(nrow + d + 4);
        union { uint4 q; __bf16 e[8]; } ur;
        ur.q = *(const uint4*)(rrow + d);
        s += n0.x * (float)ur.e[0] + n0.y * (float)ur.e[1] +
             n0.z * (float)ur.e[2] + n0.w * (float)ur.e[3] +
             n1.x * (float)ur.e[4] + n1.y * (float)ur.e[5] +
             n1.z * (float)ur.e[6] + n1.w * (float)ur.e[7];
      }
      s = s * 0.125f + __logf(conf[b * KNB + kk] + 1e-8f);  // 1/sqrt(64)=0.125
      float m = s;                                   // softmax over 16-lane half
      #pragma unroll
      for (int off = 8; off >= 1; off >>= 1) m = fmaxf(m, __shfl_xor(m, off, 32));
      const float e = __expf(s - m);
      float sum = e;
      #pragma unroll
      for (int off = 8; off >= 1; off >>= 1) sum += __shfl_xor(sum, off, 32);
      s_attn[h * KNB + kk] = e / sum;
    }
    __syncthreads();

    for (int it = 0; it < 8; ++it) {     // ctx: lanes sweep d; mean over heads
      const int d = it * 32 + lane;
      float c0 = 0.f, c1 = 0.f;
      #pragma unroll
      for (int kk = 0; kk < KNB; ++kk) {
        const float nv = s_nb[kk * EMB + d];
        c0 += s_attn[kk] * nv;
        c1 += s_attn[KNB + kk] * nv;
      }
      s_ctx[bs * EMB + d] = 0.5f * (c0 + c1);
    }
    __syncthreads();
  }

  // ---- gate = sigmoid([center|ctx] x Wg^T + b) ; fused epilogue ------------
  for (int nt = 0; nt < 16; ++nt) {
    v8f acc = {};
    const int n = nt * 16 + lm;
    #pragma unroll
    for (int ks = 0; ks < 16; ++ks) {
      ABu bb;
      v16bf av;
      if (ks < 8) {                      // gate_in cols 0..255 = center (bf16)
        const int k0 = ks * 32;
        ABu a;
        a.q[0] = *(const uint4*)(s_center + lm * EMB + k0 + half * 8);
        a.q[1] = *(const uint4*)(s_center + lm * EMB + k0 + 16 + half * 8);
        av = a.v;
      } else {                           // gate_in cols 256..511 = ctx (pack f32)
        const int k0 = (ks - 8) * 32;
        const float* c1p = s_ctx + lm * EMB + k0 + half * 8;
        const float* c2p = s_ctx + lm * EMB + k0 + 16 + half * 8;
        const float4 f0 = *(const float4*)(c1p);
        const float4 f1 = *(const float4*)(c1p + 4);
        const float4 f2 = *(const float4*)(c2p);
        const float4 f3 = *(const float4*)(c2p + 4);
        ABe a;
        a.e[0]=(__bf16)f0.x;  a.e[1]=(__bf16)f0.y;  a.e[2]=(__bf16)f0.z;  a.e[3]=(__bf16)f0.w;
        a.e[4]=(__bf16)f1.x;  a.e[5]=(__bf16)f1.y;  a.e[6]=(__bf16)f1.z;  a.e[7]=(__bf16)f1.w;
        a.e[8]=(__bf16)f2.x;  a.e[9]=(__bf16)f2.y;  a.e[10]=(__bf16)f2.z; a.e[11]=(__bf16)f2.w;
        a.e[12]=(__bf16)f3.x; a.e[13]=(__bf16)f3.y; a.e[14]=(__bf16)f3.z; a.e[15]=(__bf16)f3.w;
        av = a.v;
      }
      const int k0 = ks * 32;
      bb.q[0] = *(const uint4*)(wg_b + n * GIN + k0 + half * 8);
      bb.q[1] = *(const uint4*)(wg_b + n * GIN + k0 + 16 + half * 8);
      acc = __builtin_amdgcn_wmma_f32_16x16x32_bf16(false, av, false, bb.v,
                                                    (short)0, acc, false, false);
    }
    const float bg = b_gate[n];
    #pragma unroll
    for (int i = 0; i < 8; ++i) {
      const int  bs = i + 8 * half;
      const long b  = b0 + bs;
      const float c = center[b * EMB + n];     // exact f32 center for the mix
      float v;
      if (valid[b]) {
        const float g = 1.f / (1.f + __expf(-(acc[i] + bg)));
        v = g * c + (1.f - g) * s_ctx[bs * EMB + n];
      } else {
        v = c;
      }
      out[b * EMB + n] = v;
    }
  }
}

extern "C" void kernel_launch(void* const* d_in, const int* in_sizes, int n_in,
                              void* d_out, int out_size, void* d_ws, size_t ws_size,
                              hipStream_t stream) {
  (void)n_in; (void)out_size; (void)ws_size;
  const float*         center = (const float*)d_in[0];
  const float*         neigh  = (const float*)d_in[1];
  const float*         conf   = (const float*)d_in[2];
  const unsigned char* valid  = (const unsigned char*)d_in[3];  // bool (1B)
  const float*         Wq     = (const float*)d_in[4];
  const float*         Wk     = (const float*)d_in[5];
  const float*         Wg     = (const float*)d_in[6];
  const float*         bg     = (const float*)d_in[7];
  float*               out    = (float*)d_out;

  __bf16* wq_b = (__bf16*)d_ws;                 // 64 KB
  __bf16* wk_t = wq_b + NH * AD * EMB;          // 64 KB
  __bf16* wg_b = wk_t + NH * AD * EMB;          // 256 KB

  const int B = in_sizes[0] / EMB;              // 65536

  prep_weights<<<(EMB * GIN + 255) / 256, 256, 0, stream>>>(Wq, Wk, Wg,
                                                            wq_b, wk_t, wg_b);
  fused_attn<<<B / 16, 32, 0, stream>>>(center, neigh, conf, valid, bg,
                                        wq_b, wk_t, wg_b, out);
}
